// EGNN_2748779069775
// MI455X (gfx1250) — matrix operand order repaired and checked by reading, not compiled, not run on hardware
//
#include <hip/hip_runtime.h>

// EGNN layer for MI455X (gfx1250, wave32, WMMA).
// Layer-1 edge GEMM algebraically folded into per-node projections Q/P:
//   edge_in @ We1 = Q_i + P_j + d2 * We1[128,:]
// Dense work (h[288]->m[32], m[32]->cw[64]) on v_wmma_f32_16x16x32_f16.
// j-range split 4 ways for grid parallelism; deterministic partial sums.
// P chunk staged to LDS with global_load_async_to_lds_b128 (ASYNCcnt).

#define NODES  512
#define BATCH  4
#define BN     (BATCH * NODES)
#define DIM    64
#define MSG    32
#define EIN    129
#define HID    258          // 2*EIN
#define KP     288          // HID padded to multiple of 32 (9 K-steps)
#define KSTEPS (KP / 32)
#define RSH    296          // hbuf row stride in halves (pad for LDS banking)
#define JT     8            // j per chunk
#define PAIRS  128          // 16 i * JT j
#define NCHUNK (NODES / JT) // 64
#define SPLITS 4
#define CPS    (NCHUNK / SPLITS) // 16 chunks per block
#define MST    40           // mbuf row stride (halves)
#define QST    264          // Qs row stride (floats)
#define PST    260          // Q/P workspace row stride (floats, 16B-aligned rows)
#define PSST   264          // Ps LDS row stride (floats)
#define NBST   132          // nbuf row stride (floats)

typedef _Float16 f16;
typedef _Float16 v16h __attribute__((ext_vector_type(16)));
typedef _Float16 v8h  __attribute__((ext_vector_type(8)));
typedef float    v8f  __attribute__((ext_vector_type(8)));
typedef int      b128_t __attribute__((vector_size(16)));   // matches builtin param

#define AS1 __attribute__((address_space(1)))
#define AS3 __attribute__((address_space(3)))

#if defined(__has_builtin)
#  if __has_builtin(__builtin_amdgcn_global_load_async_to_lds_b128)
#    define HAVE_ASYNC_LDS 1
#  endif
#endif
#ifndef HAVE_ASYNC_LDS
#  define HAVE_ASYNC_LDS 0
#endif

__device__ __forceinline__ float silu_f(float x) { return x / (1.0f + __expf(-x)); }

// A-fragment loader for V_WMMA_F32_16X16X32_F16 (16-bit A 16x32), contiguous k order matching B packing:
// lanes 0-15: row M=lane, K={kb..kb+15};
// lanes 16-31: row M=lane-16, K={kb+16..kb+31}.
__device__ __forceinline__ v16h load_a16(const f16* base, int rstride, int kbase, int lane) {
  const int r  = lane & 15;
  const int hi = lane >> 4;
  const f16* p = base + r * rstride + kbase + hi * 16;
  union { v16h v; v8h h[2]; } t;
  t.h[0] = *(const v8h*)(p);
  t.h[1] = *(const v8h*)(p + 8);
  return t.v;
}

// ---------------------------------------------------------------------------
// Kernel 1: per-node projections (bias folded into Q).
// ---------------------------------------------------------------------------
__global__ void egnn_node_proj(const float* __restrict__ nf,
                               const float* __restrict__ We1,
                               const float* __restrict__ be1,
                               float* __restrict__ Qg,
                               float* __restrict__ Pg) {
  __shared__ float nfs[DIM];
  const int node = blockIdx.x;
  const int tid  = threadIdx.x;
  if (tid < DIM) nfs[tid] = nf[node * DIM + tid];
  __syncthreads();
  for (int k = tid; k < HID; k += blockDim.x) {
    float q = be1[k], p = 0.f;
    for (int d = 0; d < DIM; ++d) {
      const float x = nfs[d];
      q += x * We1[d * HID + k];
      p += x * We1[(DIM + d) * HID + k];
    }
    Qg[(size_t)node * PST + k] = q;
    Pg[(size_t)node * PST + k] = p;
  }
}

// ---------------------------------------------------------------------------
// Kernel 2: pair kernel. Block = (batch, 16-i tile, j-split). 256 thr = 8 waves.
// ---------------------------------------------------------------------------
__global__ void __launch_bounds__(256)
egnn_pairs(const float* __restrict__ coords,
           const float* __restrict__ We1,  const float* __restrict__ We2,
           const float* __restrict__ be2,  const float* __restrict__ Wc1,
           const float* __restrict__ bc1,  const float* __restrict__ Wc2,
           const float* __restrict__ bc2,  const float* __restrict__ Qg,
           const float* __restrict__ Pg,   float* __restrict__ m_part,
           float* __restrict__ c_part) {
  __shared__ float Qs[16 * QST];
  __shared__ float w128s[HID];
  __shared__ float be2s[MSG];
  __shared__ float bc1s[64];
  __shared__ float Wc2s[64];
  __shared__ float bc2s;
  __shared__ __attribute__((aligned(32))) f16 wbuf[KSTEPS * 2 * 32 * 16]; // We2 B-frags
  __shared__ __attribute__((aligned(32))) f16 c1buf[4 * 32 * 16];        // Wc1 B-frags
  __shared__ __attribute__((aligned(16))) f16 hbuf[PAIRS * RSH];
  __shared__ __attribute__((aligned(16))) f16 mbuf[PAIRS * MST];
  __shared__ __attribute__((aligned(16))) float Ps[JT * PSST];           // staged P chunk
  __shared__ float dists[PAIRS];
  __shared__ float rels[PAIRS * 4];
  __shared__ float wv[PAIRS];
  __shared__ float m_is[16 * MSG];
  __shared__ float cacc[16 * 4];
  __shared__ float coordi[16 * 4];

  const int tid   = threadIdx.x;
  const int lane  = tid & 31;
  const int wave  = tid >> 5;
  const int split = blockIdx.x % SPLITS;
  const int tile  = blockIdx.x / SPLITS;
  const int b     = tile / (NODES / 16);
  const int i0    = (tile % (NODES / 16)) * 16;
  const int nbase = b * NODES;
  const int cBeg  = split * CPS, cEnd = cBeg + CPS;

  // ------------------------- block init -------------------------
  for (int idx = tid; idx < 16 * HID; idx += 256) {
    const int il = idx / HID, k = idx - il * HID;
    Qs[il * QST + k] = Qg[(size_t)(nbase + i0 + il) * PST + k];
  }
  for (int k = tid; k < HID; k += 256) w128s[k] = We1[(2 * DIM) * HID + k];
  if (tid < MSG) be2s[tid] = be2[tid];
  if (tid < 64) { bc1s[tid] = bc1[tid]; Wc2s[tid] = Wc2[tid]; }
  if (tid == 0) bc2s = bc2[0];
  // We2 -> B fragments: lane = column, K contiguous within lane.
  for (int idx = tid; idx < KSTEPS * 2 * 32 * 16; idx += 256) {
    const int h = idx & 15, L = (idx >> 4) & 31, nt = (idx >> 9) & 1, s = idx >> 10;
    const int k = s * 32 + (L >> 4) * 16 + h;
    const int n = nt * 16 + (L & 15);
    wbuf[idx] = (k < HID) ? (f16)We2[k * MSG + n] : (f16)0.f;
  }
  for (int idx = tid; idx < 4 * 32 * 16; idx += 256) {
    const int h = idx & 15, L = (idx >> 4) & 31, nt = idx >> 9;
    const int k = (L >> 4) * 16 + h;
    const int n = nt * 16 + (L & 15);
    c1buf[idx] = (f16)Wc1[k * 64 + n];
  }
  if (tid < 48) {
    const int il = tid / 3, x = tid % 3;
    coordi[il * 4 + x] = coords[(size_t)(nbase + i0 + il) * 3 + x];
  }
  for (int idx = tid; idx < 16 * MSG; idx += 256) m_is[idx] = 0.f;
  if (tid < 64) cacc[tid] = 0.f;
  for (int idx = tid; idx < PAIRS * (RSH - HID); idx += 256) {
    const int p = idx / (RSH - HID), k = HID + idx % (RSH - HID);
    hbuf[p * RSH + k] = (f16)0.f;
  }
  __syncthreads();

  // ------------------------- j-chunk loop -------------------------
  for (int c = cBeg; c < cEnd; ++c) {
    // (a0) stage this chunk's P rows into LDS via async copy (overlaps (a))
#if HAVE_ASYNC_LDS
    for (int idx = tid; idx < JT * (PST / 4); idx += 256) {
      const int row = idx / (PST / 4), c4 = idx - row * (PST / 4);
      const float* src = Pg + (size_t)(nbase + c * JT + row) * PST + c4 * 4;
      float* dst = Ps + row * PSST + c4 * 4;
      b128_t* srcg = (b128_t*)(uintptr_t)(const void*)src;     // strip const, generic
      __builtin_amdgcn_global_load_async_to_lds_b128(
          (AS1 b128_t*)srcg, (AS3 b128_t*)(void*)dst, 0, 0);
    }
#endif
    // (a) rel coords & squared distances
    if (tid < PAIRS) {
      const int p = tid, il = p >> 3, j = c * JT + (p & (JT - 1));
      float d2 = 0.f;
      #pragma unroll
      for (int x = 0; x < 3; ++x) {
        const float r = coordi[il * 4 + x] - coords[(size_t)(nbase + j) * 3 + x];
        rels[p * 4 + x] = r;
        d2 += r * r;
      }
      dists[p] = d2;
    }
#if HAVE_ASYNC_LDS
#  if defined(__has_builtin) && __has_builtin(__builtin_amdgcn_s_wait_asynccnt)
    __builtin_amdgcn_s_wait_asynccnt(0);
#  else
    asm volatile("s_wait_asynccnt 0" ::: "memory");
#  endif
#endif
    __syncthreads();
    // (b) h = silu(Q_i + P_j + d2*w128)  (two threads per pair)
    {
      const int p = tid >> 1, half = tid & 1;
      const int il = p >> 3;
      const float d2 = dists[p];
#if HAVE_ASYNC_LDS
      const float* Pj = Ps + (p & (JT - 1)) * PSST;
#else
      const int j = c * JT + (p & (JT - 1));
      const float* Pj = Pg + (size_t)(nbase + j) * PST;
      if (c + 1 < cEnd) __builtin_prefetch(Pj + JT * PST, 0, 0);
#endif
      const float* Qi = Qs + il * QST;
      f16* hrow = hbuf + p * RSH;
      const int k0 = half ? EIN : 0, k1 = k0 + EIN;
      for (int k = k0; k < k1; ++k) {
        const float x = Qi[k] + Pj[k] + d2 * w128s[k];
        hrow[k] = (f16)silu_f(x);
      }
    }
    __syncthreads();
    // (c) GEMM1: m_ij = silu(h @ We2 + be2)  [128x288 x 288x32], wave = M-tile
    {
      const int m = wave;
      v8f acc0 = {}; v8f acc1 = {};
      const f16* abase = hbuf + m * 16 * RSH;
      #pragma unroll
      for (int s = 0; s < KSTEPS; ++s) {
        const v16h a  = load_a16(abase, RSH, s * 32, lane);
        const v16h b0 = *(const v16h*)&wbuf[((s * 2 + 0) * 32 + lane) * 16];
        const v16h b1 = *(const v16h*)&wbuf[((s * 2 + 1) * 32 + lane) * 16];
        acc0 = __builtin_amdgcn_wmma_f32_16x16x32_f16(false, a, false, b0, (short)0, acc0, false, false);
        acc1 = __builtin_amdgcn_wmma_f32_16x16x32_f16(false, a, false, b1, (short)0, acc1, false, false);
      }
      const int hi = lane >> 4, col = lane & 15;
      #pragma unroll
      for (int r = 0; r < 8; ++r) {
        const int p = m * 16 + hi * 8 + r;  // C/D layout: row = 8*(lane>>4)+vgpr
        mbuf[p * MST + col]      = (f16)silu_f(acc0[r] + be2s[col]);
        mbuf[p * MST + 16 + col] = (f16)silu_f(acc1[r] + be2s[16 + col]);
      }
    }
    __syncthreads();
    // (d) m_i += sum_j m_ij ; GEMM2 + fused cw@Wc2 cross-lane reduction
    for (int slot = tid; slot < 16 * MSG; slot += 256) {
      const int il = slot >> 5, col = slot & 31;
      float s = 0.f;
      #pragma unroll
      for (int jj = 0; jj < JT; ++jj) s += (float)mbuf[(il * JT + jj) * MST + col];
      m_is[il * MSG + col] += s;
    }
    {
      const int m = wave;
      const v16h a = load_a16(mbuf + m * 16 * MST, MST, 0, lane);
      const int hi = lane >> 4, col = lane & 15;
      float part[8];
      #pragma unroll
      for (int r = 0; r < 8; ++r) part[r] = 0.f;
      #pragma unroll
      for (int nt = 0; nt < 4; ++nt) {
        v8f acc = {};
        const v16h bf = *(const v16h*)&c1buf[(nt * 32 + lane) * 16];
        acc = __builtin_amdgcn_wmma_f32_16x16x32_f16(false, a, false, bf, (short)0, acc, false, false);
        const float wc = Wc2s[nt * 16 + col];
        const float bb = bc1s[nt * 16 + col];
        #pragma unroll
        for (int r = 0; r < 8; ++r) part[r] += silu_f(acc[r] + bb) * wc;
      }
      // reduce over the 16 lanes holding the same C rows (xor stays in half)
      #pragma unroll
      for (int r = 0; r < 8; ++r) {
        float v = part[r];
        v += __shfl_xor(v, 1);
        v += __shfl_xor(v, 2);
        v += __shfl_xor(v, 4);
        v += __shfl_xor(v, 8);
        part[r] = v;
      }
      if (col == 0) {
        #pragma unroll
        for (int r = 0; r < 8; ++r) {
          const int p = m * 16 + hi * 8 + r;
          wv[p] = (part[r] + bc2s) / fmaxf(dists[p], 1e-8f);
        }
      }
    }
    __syncthreads();
    // (e) deterministic coordinate accumulation
    if (tid < 48) {
      const int il = tid / 3, x = tid % 3;
      float s = 0.f;
      #pragma unroll
      for (int jj = 0; jj < JT; ++jj) {
        const int p = il * JT + jj;
        s += wv[p] * rels[p * 4 + x];
      }
      cacc[il * 4 + x] += s;
    }
    __syncthreads();
  }

  // ------------------------- write partials -------------------------
  for (int idx = tid; idx < 16 * MSG; idx += 256) {
    const int il = idx >> 5, col = idx & 31;
    m_part[((size_t)split * BN + nbase + i0 + il) * MSG + col] = m_is[idx];
  }
  if (tid < 48) {
    const int il = tid / 3, x = tid % 3;
    c_part[((size_t)split * BN + nbase + i0 + il) * 4 + x] = cacc[il * 4 + x];
  }
}

// ---------------------------------------------------------------------------
// Kernel 3: reduce partials, coordinate output, node MLP.
// ---------------------------------------------------------------------------
__global__ void __launch_bounds__(256)
egnn_finish(const float* __restrict__ nf,   const float* __restrict__ coords,
            const float* __restrict__ Wn1,  const float* __restrict__ bn1,
            const float* __restrict__ Wn2,  const float* __restrict__ bn2,
            const float* __restrict__ m_part, const float* __restrict__ c_part,
            float* __restrict__ node_out,   float* __restrict__ coors_out) {
  __shared__ float m_is[16 * MSG];
  __shared__ float nbuf[16 * NBST];
  const int tid   = threadIdx.x;
  const int b     = blockIdx.x / (NODES / 16);
  const int i0    = (blockIdx.x % (NODES / 16)) * 16;
  const int nbase = b * NODES;

  for (int idx = tid; idx < 16 * MSG; idx += 256) {
    const int il = idx >> 5, col = idx & 31;
    const int node = nbase + i0 + il;
    float s = 0.f;
    #pragma unroll
    for (int sp = 0; sp < SPLITS; ++sp)
      s += m_part[((size_t)sp * BN + node) * MSG + col];
    m_is[idx] = s;
  }
  if (tid < 48) {
    const int il = tid / 3, x = tid % 3;
    const int node = nbase + i0 + il;
    float s = coords[(size_t)node * 3 + x];
    #pragma unroll
    for (int sp = 0; sp < SPLITS; ++sp)
      s += c_part[((size_t)sp * BN + node) * 4 + x];
    coors_out[(size_t)node * 3 + x] = s;
  }
  __syncthreads();
  // node MLP layer 1: nh = silu([nf, m_i] @ Wn1 + bn1)
  for (int slot = tid; slot < 16 * 128; slot += 256) {
    const int il = slot >> 7, u = slot & 127;
    const float* nfr = nf + (size_t)(nbase + i0 + il) * DIM;
    float x = bn1[u];
    for (int d = 0; d < DIM; ++d)  x += nfr[d] * Wn1[d * 128 + u];
    for (int m = 0; m < MSG; ++m)  x += m_is[il * MSG + m] * Wn1[(DIM + m) * 128 + u];
    nbuf[il * NBST + u] = silu_f(x);
  }
  __syncthreads();
  // node MLP layer 2 + residual
  for (int slot = tid; slot < 16 * 64; slot += 256) {
    const int il = slot >> 6, o = slot & 63;
    const float* nfr = nf + (size_t)(nbase + i0 + il) * DIM;
    float x = nfr[o] + bn2[o];
    for (int u = 0; u < 128; ++u) x += nbuf[il * NBST + u] * Wn2[u * 64 + o];
    node_out[(size_t)(nbase + i0 + il) * DIM + o] = x;
  }
}

extern "C" void kernel_launch(void* const* d_in, const int* in_sizes, int n_in,
                              void* d_out, int out_size, void* d_ws, size_t ws_size,
                              hipStream_t stream) {
  (void)in_sizes; (void)n_in; (void)out_size; (void)ws_size;
  const float* nf     = (const float*)d_in[0];
  const float* coords = (const float*)d_in[1];
  const float* We1    = (const float*)d_in[2];
  const float* be1    = (const float*)d_in[3];
  const float* We2    = (const float*)d_in[4];
  const float* be2    = (const float*)d_in[5];
  const float* Wc1    = (const float*)d_in[6];
  const float* bc1    = (const float*)d_in[7];
  const float* Wc2    = (const float*)d_in[8];
  const float* bc2    = (const float*)d_in[9];
  const float* Wn1    = (const float*)d_in[10];
  const float* bn1    = (const float*)d_in[11];
  const float* Wn2    = (const float*)d_in[12];
  const float* bn2    = (const float*)d_in[13];

  float* node_out  = (float*)d_out;
  float* coors_out = node_out + (size_t)BN * DIM;

  float* Qg     = (float*)d_ws;                       // [BN, PST]
  float* Pg     = Qg + (size_t)BN * PST;              // [BN, PST]
  float* m_part = Pg + (size_t)BN * PST;              // [SPLITS, BN, MSG]
  float* c_part = m_part + (size_t)SPLITS * BN * MSG; // [SPLITS, BN, 4]

  egnn_node_proj<<<BN, 256, 0, stream>>>(nf, We1, be1, Qg, Pg);
  egnn_pairs<<<BATCH * (NODES / 16) * SPLITS, 256, 0, stream>>>(
      coords, We1, We2, be2, Wc1, bc1, Wc2, bc2, Qg, Pg, m_part, c_part);
  egnn_finish<<<BATCH * (NODES / 16), 256, 0, stream>>>(
      nf, coords, Wn1, bn1, Wn2, bn2, m_part, c_part, node_out, coors_out);
}
